// CBformerLayer2_46823733461568
// MI455X (gfx1250) — matrix operand order repaired and checked by reading, not compiled, not run on hardware
//
#include <hip/hip_runtime.h>
#include <hip/hip_fp16.h>

typedef __attribute__((ext_vector_type(16))) _Float16 v16h;
typedef __attribute__((ext_vector_type(8)))  float    v8f;

#define DEV __device__ __forceinline__

DEV v16h frag_load(const _Float16* p0, const _Float16* p1) {
  union { v16h v; uint4 q[2]; } u;
  u.q[0] = *(const uint4*)p0;
  u.q[1] = *(const uint4*)p1;
  return u.v;
}
DEV v8f acc_zero() { v8f z = {0.f,0.f,0.f,0.f,0.f,0.f,0.f,0.f}; return z; }
DEV v8f wmma16(v16h a, v16h b, v8f c) {
  return __builtin_amdgcn_wmma_f32_16x16x32_f16(false, a, false, b, (short)0, c, false, false);
}
DEV void lds_fence() { asm volatile("s_wait_dscnt 0" ::: "memory"); }
#define WAIT_ASYNC(n) asm volatile("s_wait_asynccnt " #n ::: "memory")

// Async DMA: global f16 blob -> LDS, 16B per lane per issue (ASYNCcnt-tracked).
DEV void stage_async(const _Float16* __restrict__ g, _Float16* l, int n128, int tid) {
  for (int i = tid; i < n128; i += 256) {
    const _Float16* gp = g + (size_t)i * 8;
    unsigned loff = (unsigned)(size_t)(l + (size_t)i * 8);
    asm volatile("global_load_async_to_lds_b128 %0, %1, off"
                 :: "v"(loff), "v"(gp) : "memory");
  }
}

// ---------------------------------------------------------------------------
// Prep: swizzle weights into per-WMMA-fragment blobs (f16).
// Conv blob: [conv(5)][kstep(18)][ntile(4)][lane(32)][i(16)]
//   co=(l&15)+16*nt, ci=(s&1)*32+(l>>4)*8+(i&7)+(i>>3)*16, tap=s>>1 -> dy,dx
// ---------------------------------------------------------------------------
__global__ void prep_wblob(const float* __restrict__ qw, const float* __restrict__ kw,
                           const float* __restrict__ vw, const float* __restrict__ c1w,
                           const float* __restrict__ c2w, _Float16* __restrict__ blob) {
  int i = blockIdx.x * 256 + threadIdx.x;
  if (i >= 5 * 18 * 4 * 32 * 16) return;
  int ii = i & 15, l = (i >> 4) & 31, nt = (i >> 9) & 3, st = i >> 11;
  int cv = st / 18, s = st - cv * 18;
  int co = (l & 15) + (nt << 4);
  int ci = ((s & 1) << 5) + ((l >> 4) << 3) + (ii & 7) + ((ii >> 3) << 4);
  int tap = s >> 1, dy = tap / 3, dx = tap % 3;
  const float* w = cv == 0 ? qw : cv == 1 ? kw : cv == 2 ? vw : cv == 3 ? c1w : c2w;
  blob[i] = (_Float16)w[((co * 64 + ci) * 3 + dy) * 3 + dx];
}

// Proj blob: [kstep(8)][ntile(16)][lane(32)][i(16)] ; B[f][e]=proj_w[e][f]
__global__ void prep_pblob(const float* __restrict__ pw, _Float16* __restrict__ blob) {
  int i = blockIdx.x * 256 + threadIdx.x; // 65536
  int ii = i & 15, l = (i >> 4) & 31, ntile = (i >> 9) & 15, s = i >> 13;
  int f = s * 32 + ((l >> 4) << 3) + (ii & 7) + ((ii >> 3) << 4);
  int e = (l & 15) + (ntile << 4);
  blob[i] = (_Float16)pw[e * 256 + f];
}

// ---------------------------------------------------------------------------
// Conv GEMM phase for one of q/k/v on the zero-padded 18x18x64 halo tile:
// s-outer (B-frag loaded once, used for both pixel rows), branch-free A loads,
// token store + LayerNorm.
// ---------------------------------------------------------------------------
DEV void conv_phase(const _Float16* xin, const _Float16* wl, _Float16* dst,
                    const float* cbias, const float* g, const float* bb,
                    int transposed, int wave, int lane) {
  const int hi = lane >> 4, ln = lane & 15;
  v8f acc[2][4];
#pragma unroll
  for (int m2 = 0; m2 < 2; ++m2)
#pragma unroll
    for (int nt = 0; nt < 4; ++nt) acc[m2][nt] = acc_zero();

  const int Y0 = wave * 2;
  for (int s = 0; s < 18; ++s) {
    int tap = s >> 1, dy = tap / 3 - 1, dx = tap % 3 - 1;
    int cib = ((s & 1) << 5) + (hi << 3);
    int sx = ln + dx + 1;                       // halo coords [0,18)
    v16h a[2];
#pragma unroll
    for (int m2 = 0; m2 < 2; ++m2) {
      int sy = Y0 + m2 + dy + 1;
      const _Float16* p = xin + (sy * 18 + sx) * 64 + cib;
      a[m2] = frag_load(p, p + 16);
    }
    const _Float16* wb = wl + ((s << 7) + lane) * 16;
#pragma unroll
    for (int nt = 0; nt < 4; ++nt) {
      const _Float16* wp = wb + nt * 512;
      v16h bf = frag_load(wp, wp + 8);
      acc[0][nt] = wmma16(a[0], bf, acc[0][nt]);
      acc[1][nt] = wmma16(a[1], bf, acc[1][nt]);
    }
  }
  // pixel(Y,X,co) -> t=(Y>>1)*8+(X>>1), e=co*4+(Y&1)*2+(X&1)
#pragma unroll
  for (int m2 = 0; m2 < 2; ++m2) {
    int Y = Y0 + m2;
#pragma unroll
    for (int nt = 0; nt < 4; ++nt) {
      int co = ln + (nt << 4);
      float cb = cbias[co];
#pragma unroll
      for (int r = 0; r < 8; ++r) {
        int X = r + (hi << 3);
        int t = ((Y >> 1) << 3) + (X >> 1);
        int e = (co << 2) + ((Y & 1) << 1) + (X & 1);
        float val = acc[m2][nt][r] + cb;
        if (transposed) dst[(e << 6) + t] = (_Float16)val;
        else            dst[(t << 8) + e] = (_Float16)val;
      }
    }
  }
  lds_fence();
  // LayerNorm over this wave's own 8 tokens
  for (int tt = 0; tt < 8; ++tt) {
    int t = (wave << 3) + tt;
    float vals[8];
    if (!transposed) {
      const _Float16* p = dst + (t << 8) + (lane << 3);
#pragma unroll
      for (int q = 0; q < 8; ++q) vals[q] = (float)p[q];
    } else {
#pragma unroll
      for (int q = 0; q < 8; ++q) vals[q] = (float)dst[((lane + (q << 5)) << 6) + t];
    }
    float sm = 0.f, sq = 0.f;
#pragma unroll
    for (int q = 0; q < 8; ++q) { sm += vals[q]; sq += vals[q] * vals[q]; }
    for (int m = 16; m >= 1; m >>= 1) { sm += __shfl_xor(sm, m, 32); sq += __shfl_xor(sq, m, 32); }
    float mean = sm * (1.f / 256.f);
    float inv = rsqrtf(sq * (1.f / 256.f) - mean * mean + 1e-5f);
    if (!transposed) {
      _Float16* p = dst + (t << 8) + (lane << 3);
#pragma unroll
      for (int q = 0; q < 8; ++q) { int e = (lane << 3) + q; p[q] = (_Float16)((vals[q] - mean) * inv * g[e] + bb[e]); }
    } else {
#pragma unroll
      for (int q = 0; q < 8; ++q) { int e = lane + (q << 5); dst[(e << 6) + t] = (_Float16)((vals[q] - mean) * inv * g[e] + bb[e]); }
    }
  }
}

// ---------------------------------------------------------------------------
// Kernel A: one WG per 16x16 window, 8 waves = 8 heads.
// LDS: xin(18x18x64) 40.5K | qs 32K | ks 32K | vsT 32K | bias 7.2K | 2x72K wbuf
// Weight blobs streamed by async DMA, double buffered, overlapped with WMMA.
// ---------------------------------------------------------------------------
__global__ __launch_bounds__(256) void win_attn_kernel(
    const float* __restrict__ x,
    const float* __restrict__ qb, const float* __restrict__ kbv, const float* __restrict__ vb,
    const float* __restrict__ gq, const float* __restrict__ bq,
    const float* __restrict__ gk, const float* __restrict__ bk,
    const float* __restrict__ gv, const float* __restrict__ bv,
    const float* __restrict__ bias_table, const float* __restrict__ proj_b,
    const _Float16* __restrict__ wblob, const _Float16* __restrict__ pblob,
    float* __restrict__ xr) {
  extern __shared__ char smem[];
  _Float16* xin = (_Float16*)smem;            // [18*18 pix][64 c], zero halo
  _Float16* qs  = xin + 18 * 18 * 64;         // [64 tok][256 e]
  _Float16* ks  = qs + 64 * 256;              // later reused as attn-out
  _Float16* vsT = ks + 64 * 256;              // [256 e][64 tok]
  float* biasLds = (float*)(vsT + 64 * 256);  // [225*8]
  _Float16* wbuf0 = (_Float16*)(biasLds + 225 * 8); // 36864 halfs
  _Float16* wbuf1 = wbuf0 + 36864;

  const int tid = threadIdx.x, wave = tid >> 5, lane = tid & 31;
  const int hi = lane >> 4, ln = lane & 15;
  const int widx = blockIdx.x;
  const int b = widx >> 8, wy = (widx >> 4) & 15, wx = widx & 15;
  const float* xwin = x + (size_t)b * 64 * 65536 + (size_t)(wy * 16) * 256 + wx * 16;

  // kick off async DMA of conv0/conv1 weight blobs (18 issues/wave each)
  stage_async(wblob,         wbuf0, 4608, tid);
  stage_async(wblob + 36864, wbuf1, 4608, tid);

  // zero-fill halo tile (b128), bias table; then interior fill after barrier
  {
    uint4 z = make_uint4(0u,0u,0u,0u);
    uint4* xz = (uint4*)xin;
    for (int i = tid; i < 18 * 18 * 64 / 8; i += 256) xz[i] = z;
  }
  for (int it = tid; it < 225 * 8; it += 256) biasLds[it] = bias_table[it];
  __syncthreads();
  {
    int Y = tid >> 4, X = tid & 15;
    _Float16* dstp = xin + ((Y + 1) * 18 + (X + 1)) * 64;
    const float* srcp = xwin + Y * 256 + X;
    for (int c = 0; c < 64; ++c) dstp[c] = (_Float16)srcp[(size_t)c * 65536];
  }

  WAIT_ASYNC(18);      // conv0 blob landed (conv1 may still fly)
  __syncthreads();
  conv_phase(xin, wbuf0, qs, qb, gq, bq, 0, wave, lane);
  __syncthreads();     // wbuf0 free
  stage_async(wblob + 2 * 36864, wbuf0, 4608, tid);  // conv2 blob
  WAIT_ASYNC(18);      // conv1 blob landed
  __syncthreads();
  conv_phase(xin, wbuf1, ks, kbv, gk, bk, 0, wave, lane);
  __syncthreads();     // wbuf1 free
  stage_async(pblob, wbuf1, 4096, tid);              // proj steps 0..3 (16 issues)
  WAIT_ASYNC(16);      // conv2 blob landed
  __syncthreads();
  conv_phase(xin, wbuf0, vsT, vb, gv, bv, 1, wave, lane);
  __syncthreads();     // tokens ready; wbuf0 free
  stage_async(pblob + 32768, wbuf0, 4096, tid);      // proj steps 4..7

  // ---- attention (wave = head), overlapped with proj-blob DMA ----
  const int h = wave;
  const _Float16* qh = qs + (h << 5);
  const _Float16* kh = ks + (h << 5);
  v16h kf[4];
#pragma unroll
  for (int nt = 0; nt < 4; ++nt) {
    const _Float16* pb = kh + (((nt << 4) + ln) << 8) + (hi << 3);
    kf[nt] = frag_load(pb, pb + 16);
  }
  v8f S[4][4];
#pragma unroll
  for (int mt = 0; mt < 4; ++mt) {
    const _Float16* pa = qh + (((mt << 4) + ln) << 8) + (hi << 3);
    v16h a = frag_load(pa, pa + 16);
#pragma unroll
    for (int nt = 0; nt < 4; ++nt) S[mt][nt] = wmma16(a, kf[nt], acc_zero());
  }
  __syncthreads();  // qs free for P scratch

  const float SCALE = 0.35355339059327373f; // (dim//heads)^-0.5
#pragma unroll
  for (int mt = 0; mt < 4; ++mt)
#pragma unroll
    for (int r = 0; r < 8; ++r) {
      int m = (mt << 4) + (hi << 3) + r, ty = m >> 3, tx = m & 7;
      float mx = -1e30f;
#pragma unroll
      for (int nt = 0; nt < 4; ++nt) {
        int n = (nt << 4) + ln, ky = n >> 3, kx = n & 7;
        int idx = (ty - ky + 7) * 15 + (tx - kx + 7);
        float sv = S[mt][nt][r] * SCALE + biasLds[idx * 8 + h];
        S[mt][nt][r] = sv;
        mx = fmaxf(mx, sv);
      }
      for (int msk = 8; msk >= 1; msk >>= 1) mx = fmaxf(mx, __shfl_xor(mx, msk, 32));
      float sum = 0.f;
#pragma unroll
      for (int nt = 0; nt < 4; ++nt) { float e = __expf(S[mt][nt][r] - mx); S[mt][nt][r] = e; sum += e; }
      for (int msk = 8; msk >= 1; msk >>= 1) sum += __shfl_xor(sum, msk, 32);
      float inv = 1.f / sum;
#pragma unroll
      for (int nt = 0; nt < 4; ++nt) S[mt][nt][r] *= inv;
    }

  // O = P @ V, K split in two halves through 4KB/head LDS scratch
  _Float16* Pc = qs + h * 2048;
  v8f O[4][2];
#pragma unroll
  for (int mt = 0; mt < 4; ++mt) { O[mt][0] = acc_zero(); O[mt][1] = acc_zero(); }
  for (int half = 0; half < 2; ++half) {
#pragma unroll
    for (int mt = 0; mt < 4; ++mt)
#pragma unroll
      for (int r = 0; r < 8; ++r) {
        int t = (mt << 4) + (hi << 3) + r;
#pragma unroll
        for (int ntl = 0; ntl < 2; ++ntl)
          Pc[t * 32 + (ntl << 4) + ln] = (_Float16)S[mt][half * 2 + ntl][r];
      }
    lds_fence();
    v16h vf[2];
#pragma unroll
    for (int ntl = 0; ntl < 2; ++ntl) {
      int e = (h << 5) + (ntl << 4) + ln;
      const _Float16* pb = vsT + (e << 6) + (half << 5) + (hi << 3);
      vf[ntl] = frag_load(pb, pb + 16);
    }
#pragma unroll
    for (int mt = 0; mt < 4; ++mt) {
      const _Float16* pa = Pc + ((mt << 4) + ln) * 32 + (hi << 3);
      v16h a = frag_load(pa, pa + 16);
      O[mt][0] = wmma16(a, vf[0], O[mt][0]);
      O[mt][1] = wmma16(a, vf[1], O[mt][1]);
    }
    lds_fence();
  }

  // gather heads -> token layout (reuse ks)
  _Float16* os = ks;
#pragma unroll
  for (int mt = 0; mt < 4; ++mt)
#pragma unroll
    for (int ntl = 0; ntl < 2; ++ntl) {
      int e = (h << 5) + (ntl << 4) + ln;
#pragma unroll
      for (int r = 0; r < 8; ++r)
        os[(((mt << 4) + (hi << 3) + r) << 8) + e] = (_Float16)O[mt][ntl][r];
    }
  WAIT_ASYNC(0);     // both proj-blob halves landed
  __syncthreads();

  // proj: s-outer so B-frags die immediately
  v8f F[4][2];
#pragma unroll
  for (int mt = 0; mt < 4; ++mt) { F[mt][0] = acc_zero(); F[mt][1] = acc_zero(); }
  for (int s = 0; s < 8; ++s) {
    const _Float16* pbase = (s < 4) ? wbuf1 + (((s << 4)) << 9)
                                    : wbuf0 + ((((s - 4) << 4)) << 9);
    v16h bf[2];
#pragma unroll
    for (int ntl = 0; ntl < 2; ++ntl) {
      const _Float16* pb = pbase + ((((wave << 1) + ntl) << 5) + lane) * 16;
      bf[ntl] = frag_load(pb, pb + 8);
    }
#pragma unroll
    for (int mt = 0; mt < 4; ++mt) {
      const _Float16* pa = os + (((mt << 4) + ln) << 8) + (s << 5) + (hi << 3);
      v16h a = frag_load(pa, pa + 16);
      F[mt][0] = wmma16(a, bf[0], F[mt][0]);
      F[mt][1] = wmma16(a, bf[1], F[mt][1]);
    }
  }

  // + proj_b, pixel-shuffle, + shortcut, window-reverse -> xr
#pragma unroll
  for (int mt = 0; mt < 4; ++mt)
#pragma unroll
    for (int ntl = 0; ntl < 2; ++ntl) {
      int e = (wave << 5) + (ntl << 4) + ln;
      int c = e >> 2, py = (e >> 1) & 1, px = e & 1;
      float pbv = proj_b[e];
#pragma unroll
      for (int r = 0; r < 8; ++r) {
        int t = (mt << 4) + (hi << 3) + r;
        int Y = ((t >> 3) << 1) + py, X = ((t & 7) << 1) + px;
        size_t gidx = ((size_t)(b * 64 + c) * 256 + (wy * 16 + Y)) * 256 + (wx * 16 + X);
        xr[gidx] = F[mt][ntl][r] + pbv + x[gidx];
      }
    }
}

// ---------------------------------------------------------------------------
// Kernel B: fused y = xr + conv2(gelu(conv1(xr))). 20x20 halo, conv1 on 18x18.
// LDS: xin 51,200B | t1 41,472B | w1+w2 blobs 147,456B (async-staged)
// ---------------------------------------------------------------------------
__global__ __launch_bounds__(256) void mlp_kernel(
    const float* __restrict__ xr, const float* __restrict__ c1b,
    const float* __restrict__ c2b, const _Float16* __restrict__ wblob,
    float* __restrict__ y) {
  extern __shared__ char smem[];
  _Float16* xin = (_Float16*)smem;            // [400 pix][64 c]
  _Float16* t1  = xin + 20 * 20 * 64;         // [324 pix][64 c]
  _Float16* w1l = t1 + 18 * 18 * 64;          // 36864 halfs
  _Float16* w2l = w1l + 36864;

  const int tid = threadIdx.x, wave = tid >> 5, lane = tid & 31;
  const int hi = lane >> 4, ln = lane & 15;
  const int widx = blockIdx.x;
  const int b = widx >> 8, tyb = (widx >> 4) & 15, txb = widx & 15;
  const float* xb = xr + (size_t)b * 64 * 65536;
  const int gy0 = tyb * 16 - 2, gx0 = txb * 16 - 2;

  stage_async(wblob + 3 * 36864, w1l, 4608, tid);
  stage_async(wblob + 4 * 36864, w2l, 4608, tid);

  for (int it = 0; it < 100; ++it) {
    int flat = it * 256 + tid;                 // 25600
    int c = flat / 400, pix = flat - c * 400;
    int yy = pix / 20, xx = pix - yy * 20;
    int gy = gy0 + yy, gx = gx0 + xx;
    float v = 0.f;
    if ((unsigned)gy < 256u && (unsigned)gx < 256u) v = xb[(size_t)c * 65536 + gy * 256 + gx];
    xin[pix * 64 + c] = (_Float16)v;
  }
  WAIT_ASYNC(0);
  __syncthreads();

  // conv1 + exact GELU on 324 pixels; s-outer, all owned m-tiles accumulated
  {
    int ownm[3], nmt = 0;
    for (int mt = wave; mt < 21; mt += 8) ownm[nmt++] = mt;
    int py_[3], px_[3];
    for (int k = 0; k < nmt; ++k) {
      int pa = ownm[k] * 16 + ln, p = pa < 324 ? pa : 323;
      py_[k] = p / 18; px_[k] = p - 18 * py_[k];
    }
    v8f acc[3][4];
#pragma unroll
    for (int k = 0; k < 3; ++k)
#pragma unroll
      for (int nt = 0; nt < 4; ++nt) acc[k][nt] = acc_zero();

    for (int s = 0; s < 18; ++s) {
      int tap = s >> 1, dy = tap / 3 - 1, dx = tap % 3 - 1;
      int cib = ((s & 1) << 5) + (hi << 3);
      const _Float16* wb = w1l + ((s << 7) + lane) * 16;
      v16h bf[4];
#pragma unroll
      for (int nt = 0; nt < 4; ++nt) { const _Float16* wp = wb + nt * 512; bf[nt] = frag_load(wp, wp + 8); }
      for (int k = 0; k < nmt; ++k) {
        const _Float16* ap = xin + ((py_[k] + 1 + dy) * 20 + (px_[k] + 1 + dx)) * 64 + cib;
        v16h a = frag_load(ap, ap + 16);
#pragma unroll
        for (int nt = 0; nt < 4; ++nt) acc[k][nt] = wmma16(a, bf[nt], acc[k][nt]);
      }
    }
    for (int k = 0; k < nmt; ++k)
#pragma unroll
      for (int nt = 0; nt < 4; ++nt) {
        int co = ln + (nt << 4);
        float cb = c1b[co];
#pragma unroll
        for (int r = 0; r < 8; ++r) {
          int pp = ownm[k] * 16 + r + (hi << 3);
          if (pp < 324) {
            float v = acc[k][nt][r] + cb;
            v = 0.5f * v * (1.f + erff(v * 0.70710678118654752f));
            t1[pp * 64 + co] = (_Float16)v;
          }
        }
      }
  }
  __syncthreads();

  // conv2 + residual; wave -> output rows {2w, 2w+1}, s-outer
  {
    v8f acc[2][4];
#pragma unroll
    for (int m2 = 0; m2 < 2; ++m2)
#pragma unroll
      for (int nt = 0; nt < 4; ++nt) acc[m2][nt] = acc_zero();
    const int Y0 = wave * 2;
    for (int s = 0; s < 18; ++s) {
      int tap = s >> 1, dy = tap / 3 - 1, dx = tap % 3 - 1;
      int cib = ((s & 1) << 5) + (hi << 3);
      const _Float16* wb = w2l + ((s << 7) + lane) * 16;
      v16h a[2];
#pragma unroll
      for (int m2 = 0; m2 < 2; ++m2) {
        const _Float16* ap = t1 + ((Y0 + m2 + 1 + dy) * 18 + (ln + 1 + dx)) * 64 + cib;
        a[m2] = frag_load(ap, ap + 16);
      }
#pragma unroll
      for (int nt = 0; nt < 4; ++nt) {
        const _Float16* wp = wb + nt * 512;
        v16h bf = frag_load(wp, wp + 8);
        acc[0][nt] = wmma16(a[0], bf, acc[0][nt]);
        acc[1][nt] = wmma16(a[1], bf, acc[1][nt]);
      }
    }
#pragma unroll
    for (int m2 = 0; m2 < 2; ++m2) {
      int Y = Y0 + m2;
#pragma unroll
      for (int nt = 0; nt < 4; ++nt) {
        int co = ln + (nt << 4);
        float cb = c2b[co];
#pragma unroll
        for (int r = 0; r < 8; ++r) {
          int X = r + (hi << 3);
          int gy = tyb * 16 + Y, gx = txb * 16 + X;
          size_t gidx = ((size_t)(b * 64 + co) * 256 + gy) * 256 + gx;
          y[gidx] = xb[(size_t)co * 65536 + gy * 256 + gx] + acc[m2][nt][r] + cb;
        }
      }
    }
  }
}

// ---------------------------------------------------------------------------
extern "C" void kernel_launch(void* const* d_in, const int* in_sizes, int n_in,
                              void* d_out, int out_size, void* d_ws, size_t ws_size,
                              hipStream_t stream) {
  const float* x   = (const float*)d_in[0];
  const float* qw  = (const float*)d_in[1];  const float* qb = (const float*)d_in[2];
  const float* kw  = (const float*)d_in[3];  const float* kb = (const float*)d_in[4];
  const float* vw  = (const float*)d_in[5];  const float* vb = (const float*)d_in[6];
  const float* gq  = (const float*)d_in[7];  const float* bq = (const float*)d_in[8];
  const float* gk  = (const float*)d_in[9];  const float* bk = (const float*)d_in[10];
  const float* gv  = (const float*)d_in[11]; const float* bv = (const float*)d_in[12];
  const float* bias_table = (const float*)d_in[13];
  const float* proj_w = (const float*)d_in[14];
  const float* proj_b = (const float*)d_in[15];
  const float* c1w = (const float*)d_in[16]; const float* c1b = (const float*)d_in[17];
  const float* c2w = (const float*)d_in[18]; const float* c2b = (const float*)d_in[19];

  char* ws = (char*)d_ws;
  float* xr = (float*)ws;                                    // 67,108,864 B
  _Float16* wblob = (_Float16*)(ws + (size_t)4 * 64 * 256 * 256 * 4);
  _Float16* pblob = (_Float16*)((char*)wblob + (size_t)5 * 36864 * 2);

  prep_wblob<<<720, 256, 0, stream>>>(qw, kw, vw, c1w, c2w, wblob);
  prep_pblob<<<256, 256, 0, stream>>>(proj_w, pblob);

  const size_t lds_a = (size_t)(18 * 18 * 64 + 3 * 16384) * 2 + 225 * 8 * 4
                     + (size_t)2 * 36864 * 2;                 // 294,432
  (void)hipFuncSetAttribute(reinterpret_cast<const void*>(win_attn_kernel),
                            hipFuncAttributeMaxDynamicSharedMemorySize, (int)lds_a);
  win_attn_kernel<<<1024, 256, lds_a, stream>>>(
      x, qb, kb, vb, gq, bq, gk, bk, gv, bv, bias_table, proj_b, wblob, pblob, xr);

  const size_t lds_b = (size_t)(25600 + 20736 + 2 * 36864) * 2; // 240,128
  (void)hipFuncSetAttribute(reinterpret_cast<const void*>(mlp_kernel),
                            hipFuncAttributeMaxDynamicSharedMemorySize, (int)lds_b);
  mlp_kernel<<<1024, 256, lds_b, stream>>>(xr, c1b, c2b, wblob, (float*)d_out);
}
